// SwitchMoE_38852274159842
// MI455X (gfx1250) — matrix operand order repaired and checked by reading, not compiled, not run on hardware
//
#include <hip/hip_runtime.h>
#include <math.h>
#include <stdint.h>

// ---------------- problem constants (from reference) ----------------
#define Bb 4
#define Tt 4096
#define Dd 1024
#define Oo 64
#define Ee 8
#define Ff 4096
#define BT (Bb*Tt)          // 16384 tokens
#define CAP 2560            // ceil(BT/E * 1.25)
#define NBIN 4096           // histogram bins for capacity selection
#define TM 32               // tokens per FFN block (2 WMMA M-tiles)
#define FC 512              // F chunk held in LDS per phase
#define XLD (Dd + 8)        // padded LDS row stride (bf16 elems)
#define HLD (FC + 8)

typedef __attribute__((ext_vector_type(16))) __bf16 v16bf;
typedef __attribute__((ext_vector_type(8)))  float  v8f;

// ---- TDM availability (probe-verified builtins; arity differs per toolchain) ----
#if defined(__has_builtin)
#  if __has_builtin(__builtin_amdgcn_tensor_load_to_lds) && \
      __has_builtin(__builtin_amdgcn_s_wait_tensorcnt)
#    define USE_TDM 1
#  endif
#endif
#ifndef USE_TDM
#  define USE_TDM 0
#endif

#if USE_TDM
typedef unsigned int tdm_v4u __attribute__((ext_vector_type(4)));
typedef int          tdm_v4i __attribute__((ext_vector_type(4)));
typedef int          tdm_v8i __attribute__((ext_vector_type(8)));
#endif

// A-fragment (16x32 bf16), contiguous K to match B: lanes0-15 row M=lane (K 0..15),
// lanes16-31 row M=lane-16 (K 16..31); kb = (lane>>4)*8. Two 16B LDS reads.
__device__ __forceinline__ v16bf lds_a_frag(const __bf16* row, int k0, int kb) {
  union { v16bf v; uint4 u[2]; } r;
  r.u[0] = *(const uint4*)(row + k0 + 2 * kb);
  r.u[1] = *(const uint4*)(row + k0 + 2 * kb + 8);
  return r.v;
}

__device__ __forceinline__ float gelu_exact(float v) {
  return 0.5f * v * (1.0f + erff(v * 0.70710678118654752f));
}

// ---------------- init kernels ----------------
__global__ __launch_bounds__(256) void moe_zero_f32(float* p, size_t n) {
  size_t i = (size_t)blockIdx.x * 256 + threadIdx.x;
  if (i < n) p[i] = 0.0f;
}

__global__ __launch_bounds__(256) void moe_init_ws(int* hist, int* cnt, int* proc,
                                                   int* tok_list, float* gate_list) {
  int i = blockIdx.x * 256 + threadIdx.x;
  if (i < Ee * NBIN) hist[i] = 0;
  if (i < Ee * 2)    cnt[i] = 0;
  if (i < BT)        proc[i] = 0;
  if (i < Ee * CAP)  { tok_list[i] = -1; gate_list[i] = 0.0f; }
}

// ---------------- router: logits, softmax, entropy, top-2, histogram ----------------
__global__ __launch_bounds__(256)
void moe_router(const float* __restrict__ tokens, const float* __restrict__ prior,
                const float* __restrict__ rw, const float* __restrict__ rb,
                float* __restrict__ out_probs, float* __restrict__ out_ent,
                int* __restrict__ top_idx, float* __restrict__ top_gate,
                float* __restrict__ top_prob, int* __restrict__ hist) {
  const int wave = threadIdx.x >> 5;
  const int lane = threadIdx.x & 31;
  const int t = blockIdx.x * 8 + wave;
  if (t >= BT) return;

  const float* x = tokens + (size_t)t * Dd;
  const float* p = prior  + (size_t)t * Oo;
  float acc[Ee];
#pragma unroll
  for (int e = 0; e < Ee; ++e) acc[e] = 0.0f;

  for (int d = lane; d < Dd; d += 32) {
    float xv = x[d];
    const float* wr = rw + (size_t)d * Ee;
#pragma unroll
    for (int e = 0; e < Ee; ++e) acc[e] += xv * wr[e];
  }
  for (int d = lane; d < Oo; d += 32) {
    float xv = p[d];
    const float* wr = rw + (size_t)(Dd + d) * Ee;
#pragma unroll
    for (int e = 0; e < Ee; ++e) acc[e] += xv * wr[e];
  }
#pragma unroll
  for (int off = 16; off > 0; off >>= 1) {
#pragma unroll
    for (int e = 0; e < Ee; ++e) acc[e] += __shfl_down(acc[e], off, 32);
  }

  if (lane == 0) {
    float mx = -1e30f;
#pragma unroll
    for (int e = 0; e < Ee; ++e) { acc[e] += rb[e]; mx = fmaxf(mx, acc[e]); }
    float pr[Ee]; float s = 0.0f;
#pragma unroll
    for (int e = 0; e < Ee; ++e) { pr[e] = __expf(acc[e] - mx); s += pr[e]; }
    float inv = 1.0f / s, ent = 0.0f;
#pragma unroll
    for (int e = 0; e < Ee; ++e) {
      pr[e] *= inv;
      ent -= pr[e] * logf(pr[e] + 1e-12f);
      out_probs[(size_t)t * Ee + e] = pr[e];
    }
    out_ent[t] = ent;

    int i0 = 0;
#pragma unroll
    for (int e = 1; e < Ee; ++e) if (pr[e] > pr[i0]) i0 = e;
    int i1 = (i0 == 0) ? 1 : 0;
#pragma unroll
    for (int e = 0; e < Ee; ++e) if (e != i0 && pr[e] > pr[i1]) i1 = e;

    float p0 = pr[i0], p1 = pr[i1];
    float den = p0 + p1 + 1e-12f;
    top_idx[t * 2 + 0] = i0;  top_idx[t * 2 + 1] = i1;
    top_gate[t * 2 + 0] = p0 / den;  top_gate[t * 2 + 1] = p1 / den;
    top_prob[t * 2 + 0] = p0;  top_prob[t * 2 + 1] = p1;
    int b0 = min(NBIN - 1, (int)(p0 * (float)NBIN));
    int b1 = min(NBIN - 1, (int)(p1 * (float)NBIN));
    atomicAdd(&hist[i0 * NBIN + b0], 1);
    atomicAdd(&hist[i1 * NBIN + b1], 1);
  }
}

// ---------------- per-expert capacity threshold from histogram ----------------
__global__ __launch_bounds__(64)
void moe_thresh(const int* __restrict__ hist, int* __restrict__ thr) {
  int e = threadIdx.x;
  if (e >= Ee) return;
  int running = 0, b_star = 0, quota = 0, above = 0, found = 0;
  for (int b = NBIN - 1; b >= 0; --b) {
    int h = hist[e * NBIN + b];
    running += h;
    if (!found && running >= CAP) {
      found = 1; b_star = b; above = running - h; quota = CAP - above;
      break;
    }
  }
  if (!found) { b_star = 0; above = running - hist[e * NBIN]; quota = CAP - above; }
  thr[e * 4 + 0] = b_star;
  thr[e * 4 + 1] = quota;
  thr[e * 4 + 2] = above;
  thr[e * 4 + 3] = 0;
}

// ---------------- slot assignment (capacity-limited, highest-prob kept) ----------------
__global__ __launch_bounds__(256)
void moe_assign(const int* __restrict__ top_idx, const float* __restrict__ top_gate,
                const float* __restrict__ top_prob, const int* __restrict__ thr,
                int* __restrict__ cnt, int* __restrict__ tok_list,
                float* __restrict__ gate_list, int* __restrict__ proc) {
  int t = blockIdx.x * 256 + threadIdx.x;
  if (t >= BT) return;
#pragma unroll
  for (int k = 0; k < 2; ++k) {
    int e = top_idx[t * 2 + k];
    float p = top_prob[t * 2 + k];
    int bin = min(NBIN - 1, (int)(p * (float)NBIN));
    int b_star = thr[e * 4 + 0], quota = thr[e * 4 + 1], above = thr[e * 4 + 2];
    int slot = -1;
    if (bin > b_star) {
      slot = atomicAdd(&cnt[e * 2 + 0], 1);
    } else if (bin == b_star) {
      int pos = atomicAdd(&cnt[e * 2 + 1], 1);
      if (pos < quota) slot = above + pos;
    }
    if (slot >= 0 && slot < CAP) {
      tok_list[e * CAP + slot]  = t;
      gate_list[e * CAP + slot] = top_gate[t * 2 + k];
      atomicAdd(&proc[t], 1);
    }
  }
}

// ---------------- fused expert FFN: y = gelu(x@W1+b1)@W2+b2, WMMA bf16 ----------------
// Block = 32 tokens (2 M-tiles), 8 waves. x-tile gathered by the Tensor Data Mover
// (gather-mode D#, 16-bit row indices) into LDS, then converted to bf16.
__global__ __launch_bounds__(256)
void moe_ffn(const float* __restrict__ flat,
             const float* __restrict__ w1, const float* __restrict__ b1,
             const float* __restrict__ w2, const float* __restrict__ b2,
             const int* __restrict__ tok_list, const float* __restrict__ gate_list,
             float* __restrict__ out_acc) {
  __shared__ __align__(16) __bf16 xs[TM * XLD];   // 32 x 1024 x-tile (bf16) ~66KB
  __shared__ __align__(16) __bf16 hs[TM * HLD];   // 32 x 512 h-chunk (bf16) ~33KB
  __shared__ int   s_tok[TM];
  __shared__ float s_gate[TM];

  const int tid  = threadIdx.x;
  const int wave = tid >> 5;
  const int lane = tid & 31;
  const int e    = blockIdx.x / (CAP / TM);
  const int tile = blockIdx.x % (CAP / TM);

  if (tid < TM) {
    s_tok[tid]  = tok_list[e * CAP + tile * TM + tid];
    s_gate[tid] = gate_list[e * CAP + tile * TM + tid];
  }
  __syncthreads();

#if USE_TDM
  // ---- TDM gather staging: 4 groups of 8 token rows (f32) -> hs alias -> bf16 xs ----
  {
    float* xf = (float*)hs;   // 8 rows * 1024 f32 = 32KB staging inside hs
    for (int g = 0; g < TM / 8; ++g) {
      if (tid == 0) {
        unsigned long long ga = (unsigned long long)(uintptr_t)flat;
        unsigned lds_off = (unsigned)(uintptr_t)(void*)xf;   // low 32b of generic = LDS addr
        // D# group0: count=1, gather_mode=1, 16-bit indices, lds_addr, global_addr, type=2
        tdm_v4u g0 = { 0x80000001u, lds_off,
                       (unsigned)(ga & 0xffffffffu),
                       (unsigned)((ga >> 32) & 0x1ffffffu) | (2u << 30) };
        // D# group1: data_size=4B; tensor_dim0=1024; tensor_dim1=BT; tile_dim0=1024;
        //            tile_dim1=8 valid indices; tensor_dim0_stride=1024
        tdm_v8i g1 = { 0x00020000, 0x04000000, 0x40000000, 0x04000000,
                       8, 1024, 0, 0 };
        int idx[8];
#pragma unroll
        for (int r = 0; r < 8; ++r) {
          int tk = s_tok[g * 8 + r];
          idx[r] = tk < 0 ? 0 : tk;
        }
        tdm_v4i g2 = { idx[0] | (idx[1] << 16), idx[2] | (idx[3] << 16),
                       idx[4] | (idx[5] << 16), idx[6] | (idx[7] << 16) };
        tdm_v4i g3 = { 0, 0, 0, 0 };
#if __clang_major__ >= 23
        __builtin_amdgcn_tensor_load_to_lds(g0, g1, g2, g3, (tdm_v8i)0, 0);
#else
        __builtin_amdgcn_tensor_load_to_lds(g0, g1, g2, g3, 0);
#endif
        __builtin_amdgcn_s_wait_tensorcnt(0);
      }
      __syncthreads();
      for (int i = tid; i < 8 * Dd; i += 256) {
        int r = i >> 10, d = i & (Dd - 1);
        int m = g * 8 + r;
        float v = (s_tok[m] >= 0) ? xf[i] : 0.0f;
        xs[m * XLD + d] = (__bf16)v;
      }
      __syncthreads();
    }
  }
#else
  // ---- fallback staging: vector loads + convert ----
  for (int i = tid; i < TM * Dd; i += 256) {
    int m = i >> 10, d = i & (Dd - 1);
    int tok = s_tok[m];
    float v = (tok >= 0) ? flat[(size_t)tok * Dd + d] : 0.0f;
    xs[m * XLD + d] = (__bf16)v;
  }
  __syncthreads();
#endif

  const float* W1 = w1 + (size_t)e * Dd * Ff;
  const float* W2 = w2 + (size_t)e * Ff * Dd;
  const float* B1 = b1 + (size_t)e * Ff;
  const float* B2 = b2 + (size_t)e * Dd;

  const int mrow = lane & 15;          // A fragment row (within M-tile)
  const int kb   = (lane >> 4) * 8;    // half the A fragment K base (K base = 2*kb)
  const int nlan = lane & 15;          // B fragment column / C column
  const int kh   = (lane >> 4) * 16;   // B fragment K half
  const int mofs = (lane >> 4) * 8;    // C/D fragment row offset

  v8f yacc[16];                        // [mt*8 + nt]
#pragma unroll
  for (int i = 0; i < 16; ++i) yacc[i] = (v8f){0,0,0,0,0,0,0,0};

  for (int fc0 = 0; fc0 < Ff; fc0 += FC) {
    // ---- phase 1: h_chunk = gelu(x @ W1[:, fc0:fc0+FC] + b1), wave owns 64 cols ----
    v8f hacc[8];                       // [mt*4 + nt]
#pragma unroll
    for (int i = 0; i < 8; ++i) hacc[i] = (v8f){0,0,0,0,0,0,0,0};

    for (int k0 = 0; k0 < Dd; k0 += 32) {
      v16bf a0 = lds_a_frag(xs + mrow * XLD, k0, kb);
      v16bf a1 = lds_a_frag(xs + (16 + mrow) * XLD, k0, kb);
      // prefetch next K-block of W1 into L2 while these WMMAs run
      if (k0 + 32 < Dd)
        __builtin_prefetch(W1 + (size_t)(k0 + 32 + kh + nlan) * Ff + fc0 + wave * 64, 0, 1);
#pragma unroll
      for (int nt = 0; nt < 4; ++nt) {
        int ncol = fc0 + wave * 64 + nt * 16 + nlan;
        const float* wp = W1 + (size_t)(k0 + kh) * Ff + ncol;
        v16bf bfr;
#pragma unroll
        for (int j = 0; j < 16; ++j) bfr[j] = (__bf16)wp[(size_t)j * Ff];
        hacc[nt] = __builtin_amdgcn_wmma_f32_16x16x32_bf16(
            false, a0, false, bfr, (short)0, hacc[nt], false, false);
        hacc[4 + nt] = __builtin_amdgcn_wmma_f32_16x16x32_bf16(
            false, a1, false, bfr, (short)0, hacc[4 + nt], false, false);
      }
    }
#pragma unroll
    for (int mt = 0; mt < 2; ++mt) {
#pragma unroll
      for (int nt = 0; nt < 4; ++nt) {
        int nc = wave * 64 + nt * 16 + nlan;
        float bias = B1[fc0 + nc];
#pragma unroll
        for (int r = 0; r < 8; ++r) {
          float v = hacc[mt * 4 + nt][r] + bias;
          hs[(mt * 16 + r + mofs) * HLD + nc] = (__bf16)gelu_exact(v);
        }
      }
    }
    __syncthreads();

    // ---- phase 2: y += h_chunk @ W2[fc0:fc0+FC, :], wave owns 128 cols of D ----
    for (int k0 = 0; k0 < FC; k0 += 32) {
      v16bf a0 = lds_a_frag(hs + mrow * HLD, k0, kb);
      v16bf a1 = lds_a_frag(hs + (16 + mrow) * HLD, k0, kb);
      if (k0 + 32 < FC)
        __builtin_prefetch(W2 + (size_t)(fc0 + k0 + 32 + kh + nlan) * Dd + wave * 128, 0, 1);
#pragma unroll
      for (int nt = 0; nt < 8; ++nt) {
        int ncol = wave * 128 + nt * 16 + nlan;
        const float* wp = W2 + (size_t)(fc0 + k0 + kh) * Dd + ncol;
        v16bf bfr;
#pragma unroll
        for (int j = 0; j < 16; ++j) bfr[j] = (__bf16)wp[(size_t)j * Dd];
        yacc[nt] = __builtin_amdgcn_wmma_f32_16x16x32_bf16(
            false, a0, false, bfr, (short)0, yacc[nt], false, false);
        yacc[8 + nt] = __builtin_amdgcn_wmma_f32_16x16x32_bf16(
            false, a1, false, bfr, (short)0, yacc[8 + nt], false, false);
      }
    }
    __syncthreads();   // hs reused next chunk
  }

  // ---- epilogue: bias + gate + scatter-add ----
#pragma unroll
  for (int mt = 0; mt < 2; ++mt) {
#pragma unroll
    for (int nt = 0; nt < 8; ++nt) {
      int ncol = wave * 128 + nt * 16 + nlan;
      float bias = B2[ncol];
#pragma unroll
      for (int r = 0; r < 8; ++r) {
        int m = mt * 16 + r + mofs;
        int tok = s_tok[m];
        if (tok >= 0) {
          float v = (yacc[mt * 8 + nt][r] + bias) * s_gate[m];
          atomicAdd(out_acc + (size_t)tok * Dd + ncol, v);
        }
      }
    }
  }
}

// ---------------- passthrough for dropped tokens ----------------
__global__ __launch_bounds__(256)
void moe_finalize(const float* __restrict__ flat, const int* __restrict__ proc,
                  float* __restrict__ out_acc) {
  size_t i = (size_t)blockIdx.x * 256 + threadIdx.x;
  if (i >= (size_t)BT * Dd) return;
  int t = (int)(i >> 10);
  if (proc[t] == 0) out_acc[i] = flat[i];
}

// ---------------- host launcher ----------------
extern "C" void kernel_launch(void* const* d_in, const int* in_sizes, int n_in,
                              void* d_out, int out_size, void* d_ws, size_t ws_size,
                              hipStream_t stream) {
  (void)in_sizes; (void)n_in; (void)out_size; (void)ws_size;
  const float* tokens = (const float*)d_in[0];
  const float* prior  = (const float*)d_in[1];
  const float* rw     = (const float*)d_in[2];
  const float* rb     = (const float*)d_in[3];
  const float* w1     = (const float*)d_in[4];
  const float* b1     = (const float*)d_in[5];
  const float* w2     = (const float*)d_in[6];
  const float* b2     = (const float*)d_in[7];

  float* out_acc   = (float*)d_out;                    // (BT, D)
  float* out_probs = out_acc + (size_t)BT * Dd;        // (BT, E)
  float* out_ent   = out_probs + (size_t)BT * Ee;      // (BT,)

  char* w = (char*)d_ws;
  int*   top_idx   = (int*)w;    w += (size_t)BT * 2 * sizeof(int);
  float* top_gate  = (float*)w;  w += (size_t)BT * 2 * sizeof(float);
  float* top_prob  = (float*)w;  w += (size_t)BT * 2 * sizeof(float);
  int*   hist      = (int*)w;    w += (size_t)Ee * NBIN * sizeof(int);
  int*   thr       = (int*)w;    w += (size_t)Ee * 4 * sizeof(int);
  int*   cnt       = (int*)w;    w += (size_t)Ee * 2 * sizeof(int);
  int*   tok_list  = (int*)w;    w += (size_t)Ee * CAP * sizeof(int);
  float* gate_list = (float*)w;  w += (size_t)Ee * CAP * sizeof(float);
  int*   proc      = (int*)w;    w += (size_t)BT * sizeof(int);

  const size_t nOut = (size_t)BT * Dd;
  moe_zero_f32<<<(int)((nOut + 255) / 256), 256, 0, stream>>>(out_acc, nOut);
  moe_init_ws<<<(Ee * NBIN + 255) / 256, 256, 0, stream>>>(hist, cnt, proc, tok_list, gate_list);
  moe_router<<<BT / 8, 256, 0, stream>>>(tokens, prior, rw, rb, out_probs, out_ent,
                                         top_idx, top_gate, top_prob, hist);
  moe_thresh<<<1, 64, 0, stream>>>(hist, thr);
  moe_assign<<<BT / 256, 256, 0, stream>>>(top_idx, top_gate, top_prob, thr, cnt,
                                           tok_list, gate_list, proc);
  moe_ffn<<<Ee * (CAP / TM), 256, 0, stream>>>(tokens, w1, b1, w2, b2,
                                               tok_list, gate_list, out_acc);
  moe_finalize<<<(int)((nOut + 255) / 256), 256, 0, stream>>>(tokens, proc, out_acc);
}
